// _RPN_66116726555444
// MI455X (gfx1250) — matrix operand order repaired and checked by reading, not compiled, not run on hardware
//
#include <hip/hip_runtime.h>
#include <hip/hip_bf16.h>

typedef __attribute__((ext_vector_type(16))) __bf16 v16bf;
typedef __attribute__((ext_vector_type(8)))  __bf16 v8bf;
typedef __attribute__((ext_vector_type(8)))  float  v8f;
typedef __attribute__((ext_vector_type(4)))  unsigned int v4u;
typedef __attribute__((ext_vector_type(8)))  int v8i;
typedef __attribute__((ext_vector_type(4)))  int v4i;

#define HH_ 24
#define WW_ 24
#define HW_ 576
#define SD_ 16
#define DIN_ 256

// ---- helpers: manual bf16 <-> f32 (round-to-nearest-even) ----
__device__ __forceinline__ unsigned short f2bf(float f){
  unsigned u = __float_as_uint(f);
  u += 0x7FFFu + ((u >> 16) & 1u);
  return (unsigned short)(u >> 16);
}
__device__ __forceinline__ float bf_lo(unsigned u){ return __uint_as_float(u << 16); }
__device__ __forceinline__ float bf_hi(unsigned u){ return __uint_as_float(u & 0xFFFF0000u); }

// ---------------- pre-pass casts ----------------
// feat [B][C][D][H][W] f32 -> featT [B][D][H][W][C] bf16
__global__ __launch_bounds__(256) void k_cast_feat(const float* __restrict__ feat,
                                                   unsigned short* __restrict__ featT){
  int tid = blockIdx.x * 256 + threadIdx.x;          // 4,718,592
  int c = tid & 255, m = tid >> 8;
  int hw = m % HW_, dm = m / HW_;
  int d = dm & 15, b = dm >> 4;
  featT[tid] = f2bf(feat[((b * DIN_ + c) * SD_ + d) * HW_ + hw]);
}

// W_conv [O][I][27] f32 -> Wt [tap][O][I] bf16
__global__ __launch_bounds__(256) void k_cast_wconv(const float* __restrict__ Wc,
                                                    unsigned short* __restrict__ Wt){
  int tid = blockIdx.x * 256 + threadIdx.x;          // 1,769,472
  int tap = tid >> 16;
  int rem = tid & 65535;
  int o = rem >> 8, i = rem & 255;
  Wt[tid] = f2bf(Wc[(o * DIN_ + i) * 27 + tap]);
}

// bbox weights (already [N][K] row-major) -> bf16, concatenated k16|k12|k8
__global__ __launch_bounds__(256) void k_cast_wb(const float* __restrict__ w0,
                                                 const float* __restrict__ w1,
                                                 const float* __restrict__ w2,
                                                 unsigned short* __restrict__ Wb){
  int tid = blockIdx.x * 256 + threadIdx.x;          // 110,592
  float v;
  if      (tid < 49152) v = w0[tid];
  else if (tid < 86016) v = w1[tid - 49152];
  else                  v = w2[tid - 86016];
  Wb[tid] = f2bf(v);
}

// ---------------- conv3d 3x3x3 as 27 shifted GEMMs ----------------
// Block: 256 thr = 8 waves; block tile 64(M) x 256(N); wave tile 16(M) x 128(N).
// Weight panels (256 N x 32 K bf16) DMA'd to LDS by the Tensor Data Mover,
// one 32-row slice per wave, double-buffered on TENSORcnt. Rows padded to 80B
// (pad_interval = 16 DW, pad_amount = 4 DW) so every row is 16B-aligned.
// 18432 M rows / 64 = 288 blocks exact; no divergent exit around WMMA/barriers.
__global__ __launch_bounds__(256) void k_conv_wmma(const unsigned short* __restrict__ featT,
                                                   const unsigned short* __restrict__ Wt,
                                                   const float* __restrict__ bconv,
                                                   unsigned short* __restrict__ xT){
  __shared__ alignas(32) unsigned short sB[2][10240];   // 2 x (256 rows x 40 ushort) = 40,960 B

  int wv   = threadIdx.x >> 5;       // 0..7
  int lane = threadIdx.x & 31;
  int g = lane >> 4, r = lane & 15;
  int mi = wv >> 1;                  // 0..3 : M sub-tile
  int nh = wv & 1;                   // 0..1 : N half (128 cols)

  int m = blockIdx.x * 64 + mi * 16 + r;   // A-fragment row for this lane
  int hw = m % HW_, dm = m / HW_;
  int d = dm & 15, b = dm >> 4;
  int h = hw / WW_, w = hw % WW_;

  // --- TDM descriptor pieces (ISA 8.3/8.4) ---
  int rowStart = wv * 32;            // this wave's 32-row slice of the N=256 panel
  unsigned long long wtA = (unsigned long long)(uintptr_t)Wt;
  unsigned ldsSlice[2] = {
    (unsigned)(uintptr_t)&sB[0][0] + (unsigned)rowStart * 80u,
    (unsigned)(uintptr_t)&sB[1][0] + (unsigned)rowStart * 80u };
  // group1: data_size=2B(1), pad_enable, pad_interval=16DW(3), pad_amount=4DW(3),
  //         tensor_dim0=32, tensor_dim1=256, tile_dim0=32, tile_dim1=32, dim0_stride=256
  const v8i g1 = { (int)((1u << 16) | (1u << 20) | (3u << 22) | (3u << 25)),
                   (int)(32u << 16),        // tensor_dim0 lo16 @ [63:48]
                   (int)(256u << 16),       // tensor_dim1 lo16 @ [111:80]->w2 hi
                   (int)(32u << 16),        // tile_dim0 @ [127:112]
                   32,                      // tile_dim1 @ [143:128]
                   256,                     // tensor_dim0_stride lo32
                   0, 0 };
  const v4i zz4 = { 0, 0, 0, 0 };
  const v8i zz8 = { 0, 0, 0, 0, 0, 0, 0, 0 };

  auto issue = [&](int sn){
    int tap = sn >> 3, kt = (sn & 7) << 5;
    unsigned long long ga = wtA + 2ull * (unsigned long long)((tap << 16) + (rowStart << 8) + kt);
    v4u g0 = { 1u,                                   // count=1 (valid user D#)
               ldsSlice[sn & 1],                     // lds_addr
               (unsigned)(ga & 0xFFFFFFFFull),       // global_addr[31:0]
               (unsigned)((ga >> 32) & 0x1FFFFFFull) | (2u << 30) };  // addr[56:32] | type=2
    __builtin_amdgcn_tensor_load_to_lds(g0, g1, zz4, zz4, zz8, 0);
  };

  v8f acc[8];
  #pragma unroll
  for (int q = 0; q < 8; ++q) acc[q] = (v8f){};

  issue(0);                                          // prologue: panel 0 in flight

  #pragma unroll 1
  for (int step = 0; step < 216; ++step){            // 27 taps x 8 K-steps
    int tap = step >> 3, kt = (step & 7) << 5;

    __syncthreads();                                 // all reads of next-buf (from step-1) done
    if (step < 215){
      issue(step + 1);                               // overlap next DMA with this compute
      __builtin_amdgcn_s_wait_tensorcnt((short)1);   // current panel slice landed
    } else {
      __builtin_amdgcn_s_wait_tensorcnt((short)0);
    }
    __syncthreads();                                 // all 8 slices of current panel landed

    // A fragment (16x32 bf16) direct from global, halo-zeroed per lane
    int kd = tap / 9, kh = (tap / 3) % 3, kw = tap % 3;
    int dd = d + kd - 1, hh = h + kh - 1, ww = w + kw - 1;
    bool valid = ((unsigned)dd < 16u) & ((unsigned)hh < 24u) & ((unsigned)ww < 24u);
    const unsigned short* aPtr =
        featT + (size_t)((((b * SD_ + dd) * HH_ + hh) * WW_ + ww)) * DIN_ + kt;
    v8bf alo = (v8bf){}, ahi = (v8bf){};
    if (valid){
      alo = *reinterpret_cast<const v8bf*>(aPtr + 8 * g);
      ahi = *reinterpret_cast<const v8bf*>(aPtr + 8 * g + 16);
    }
    v16bf a = __builtin_shufflevector(alo, ahi, 0,1,2,3,4,5,6,7,8,9,10,11,12,13,14,15);

    const unsigned short* bbase = &sB[step & 1][0];
    #pragma unroll
    for (int q = 0; q < 8; ++q){
      int n = nh * 128 + q * 16 + r;                 // B fragment: N = lane%16, K = 8g+e | 16+8g+e
      const unsigned short* bp = bbase + n * 40 + 8 * g;
      v8bf b0 = *reinterpret_cast<const v8bf*>(bp);      // ds_load_b128
      v8bf b1 = *reinterpret_cast<const v8bf*>(bp + 16); // ds_load_b128
      v16bf bb = __builtin_shufflevector(b0, b1, 0,1,2,3,4,5,6,7,8,9,10,11,12,13,14,15);
      acc[q] = __builtin_amdgcn_wmma_f32_16x16x32_bf16(false, a, false, bb, (short)0, acc[q], false, false);
    }
  }

  // epilogue: bias + relu -> bf16 channels-last x
  #pragma unroll
  for (int q = 0; q < 8; ++q){
    int n = nh * 128 + q * 16 + r;                   // D: N = lane%16
    float bias = bconv[n];
    #pragma unroll
    for (int rr = 0; rr < 8; ++rr){                  // D: M = rr + 8g
      int row = blockIdx.x * 64 + mi * 16 + rr + 8 * g;
      float v = acc[q][rr] + bias;
      v = v > 0.0f ? v : 0.0f;
      xT[(size_t)row * DIN_ + n] = f2bf(v);
    }
  }
}

// ---------------- time avg-pool: x[b,d,hw,c] -> pooled[pos,c], pos = k16|k12|k8 ----------------
__global__ __launch_bounds__(256) void k_pool(const unsigned short* __restrict__ xT,
                                              unsigned short* __restrict__ pooled){
  int tid = blockIdx.x * 256 + threadIdx.x;
  if (tid >= 552960) return;              // 17280 positions * 32 chunks
  int cc = (tid & 31) * 8;
  int pos = tid >> 5;
  int k, t, ml;
  if      (pos < 1152){ k = 16; t = 1; ml = pos; }
  else if (pos < 6912){ k = 12; t = 5; ml = pos - 1152; }
  else                { k = 8;  t = 9; ml = pos - 6912; }
  int hw = ml % HW_, bt = ml / HW_;
  int ti = bt % t, b = bt / t;
  float s[8];
  #pragma unroll
  for (int e = 0; e < 8; ++e) s[e] = 0.0f;
  const unsigned short* base = xT + (size_t)((b * SD_ + ti) * HW_ + hw) * DIN_ + cc;
  for (int j = 0; j < k; ++j){
    uint4 v = *reinterpret_cast<const uint4*>(base + (size_t)j * HW_ * DIN_);
    s[0] += bf_lo(v.x); s[1] += bf_hi(v.x);
    s[2] += bf_lo(v.y); s[3] += bf_hi(v.y);
    s[4] += bf_lo(v.z); s[5] += bf_hi(v.z);
    s[6] += bf_lo(v.w); s[7] += bf_hi(v.w);
  }
  float inv = 1.0f / (float)k;
  uint4 o;
  o.x = (unsigned)f2bf(s[0] * inv) | ((unsigned)f2bf(s[1] * inv) << 16);
  o.y = (unsigned)f2bf(s[2] * inv) | ((unsigned)f2bf(s[3] * inv) << 16);
  o.z = (unsigned)f2bf(s[4] * inv) | ((unsigned)f2bf(s[5] * inv) << 16);
  o.w = (unsigned)f2bf(s[6] * inv) | ((unsigned)f2bf(s[7] * inv) << 16);
  *reinterpret_cast<uint4*>(pooled + (size_t)pos * DIN_ + cc) = o;
}

// ---------------- cls head (N=6) + pairwise softmax + transposed write ----------------
__global__ __launch_bounds__(256) void k_cls(const unsigned short* __restrict__ pooled,
                                             const float* __restrict__ Wcls,
                                             const float* __restrict__ bcls,
                                             float* __restrict__ out){
  int pos = blockIdx.x * 256 + threadIdx.x;
  if (pos >= 17280) return;
  int probOff;
  if      (pos < 1152) probOff = 0      + pos * 6;
  else if (pos < 6912) probOff = 6912   + (pos - 1152) * 6;
  else                 probOff = 41472  + (pos - 6912) * 6;

  float s[6];
  #pragma unroll
  for (int j = 0; j < 6; ++j) s[j] = bcls[j];
  const unsigned short* a = pooled + (size_t)pos * DIN_;
  for (int c = 0; c < 256; c += 8){
    uint4 v = *reinterpret_cast<const uint4*>(a + c);
    float av[8] = { bf_lo(v.x), bf_hi(v.x), bf_lo(v.y), bf_hi(v.y),
                    bf_lo(v.z), bf_hi(v.z), bf_lo(v.w), bf_hi(v.w) };
    #pragma unroll
    for (int j = 0; j < 6; ++j){
      const float* wrow = Wcls + j * 256 + c;
      #pragma unroll
      for (int e = 0; e < 8; ++e) s[j] += av[e] * wrow[e];
    }
  }
  // softmax over pairs (c, c+3) from the (6,t)->(2,3t) reshape
  #pragma unroll
  for (int rr = 0; rr < 3; ++rr){
    float x0 = s[rr], x1 = s[rr + 3];
    float mx = fmaxf(x0, x1);
    float e0 = __expf(x0 - mx), e1 = __expf(x1 - mx);
    float inv = 1.0f / (e0 + e1);
    out[probOff + rr]     = e0 * inv;
    out[probOff + rr + 3] = e1 * inv;
  }
}

// ---------------- bbox heads as 16x16 WMMA tiles ----------------
// jobs: k16: 72*12 = 864 ; k12: 360*9 = 3240 ; k8: 648*6 = 3888 ; total 7992
// 4 waves/block -> 1998 blocks exact.
__global__ __launch_bounds__(128) void k_bbox_wmma(const unsigned short* __restrict__ pooled,
                                                   const unsigned short* __restrict__ Wb,
                                                   const float* __restrict__ bias16,
                                                   const float* __restrict__ bias12,
                                                   const float* __restrict__ bias8,
                                                   float* __restrict__ out){
  int wave = (blockIdx.x << 2) + (threadIdx.x >> 5);
  int lane = threadIdx.x & 31;
  int g = lane >> 4, r = lane & 15;

  int j, ntiles, NCH, outOff;
  const unsigned short* A;
  const unsigned short* B;
  const float* bias;
  if (wave < 864)      { j = wave;        ntiles = 12; NCH = 192; A = pooled;              B = Wb;         bias = bias16; outOff = 103680;  }
  else if (wave < 4104){ j = wave - 864;  ntiles = 9;  NCH = 144; A = pooled + 1152 * 256; B = Wb + 49152; bias = bias12; outOff = 324864;  }
  else                 { j = wave - 4104; ntiles = 6;  NCH = 96;  A = pooled + 6912 * 256; B = Wb + 86016; bias = bias8;  outOff = 1154304; }
  int mtile = j / ntiles, ntile = j % ntiles;

  const unsigned short* aPtr = A + (size_t)(mtile * 16 + r) * DIN_;
  const unsigned short* bPtr = B + (size_t)(ntile * 16 + r) * DIN_;
  v8f acc = (v8f){};
  #pragma unroll
  for (int kt = 0; kt < 256; kt += 32){
    v8bf alo = *reinterpret_cast<const v8bf*>(aPtr + kt + 16 * g);
    v8bf ahi = *reinterpret_cast<const v8bf*>(aPtr + kt + 16 * g + 8);
    v16bf a = __builtin_shufflevector(alo, ahi, 0,1,2,3,4,5,6,7,8,9,10,11,12,13,14,15);
    v16bf bb = *reinterpret_cast<const v16bf*>(bPtr + kt + 16 * g);
    acc = __builtin_amdgcn_wmma_f32_16x16x32_bf16(false, a, false, bb, (short)0, acc, false, false);
  }
  int n = ntile * 16 + r;
  float bv = bias[n];
  int m0 = mtile * 16 + 8 * g;
  #pragma unroll
  for (int rr = 0; rr < 8; ++rr)
    out[outOff + (size_t)(m0 + rr) * NCH + n] = acc[rr] + bv;
}

extern "C" void kernel_launch(void* const* d_in, const int* in_sizes, int n_in,
                              void* d_out, int out_size, void* d_ws, size_t ws_size,
                              hipStream_t stream) {
  const float* feat  = (const float*)d_in[0];
  const float* Wconv = (const float*)d_in[1];
  const float* bconv = (const float*)d_in[2];
  const float* Wcls  = (const float*)d_in[3];
  const float* bcls  = (const float*)d_in[4];
  const float* Wb16  = (const float*)d_in[5];
  const float* bb16  = (const float*)d_in[6];
  const float* Wb12  = (const float*)d_in[7];
  const float* bb12  = (const float*)d_in[8];
  const float* Wb8   = (const float*)d_in[9];
  const float* bb8   = (const float*)d_in[10];
  float* out = (float*)d_out;

  char* ws = (char*)d_ws;
  unsigned short* featT  = (unsigned short*)(ws + 0);          //  9,437,184 B
  unsigned short* Wt     = (unsigned short*)(ws + 9437184);    //  3,538,944 B
  unsigned short* WbAll  = (unsigned short*)(ws + 12976128);   //    221,184 B
  unsigned short* xT     = (unsigned short*)(ws + 13197312);   //  9,437,184 B
  unsigned short* pooled = (unsigned short*)(ws + 22634496);   //  8,847,360 B  (total ~31.5 MB)

  k_cast_feat <<<18432, 256, 0, stream>>>(feat, featT);
  k_cast_wconv<<< 6912, 256, 0, stream>>>(Wconv, Wt);
  k_cast_wb   <<<  432, 256, 0, stream>>>(Wb16, Wb12, Wb8, WbAll);
  k_conv_wmma <<<  288, 256, 0, stream>>>(featT, Wt, bconv, xT);
  k_pool      <<< 2160, 256, 0, stream>>>(xT, pooled);
  k_cls       <<<   68, 256, 0, stream>>>(pooled, Wcls, bcls, out);
  k_bbox_wmma <<< 1998, 128, 0, stream>>>(pooled, WbAll, bb16, bb12, bb8, out);
}
